// InterpAttentionKHeadsNet_21345987461165
// MI455X (gfx1250) — compile-verified, matrix operand order checked by reading, not run on hardware
//
#include <hip/hip_runtime.h>
#include <hip/hip_bf16.h>

typedef __attribute__((ext_vector_type(16))) _Float16 v16h;
typedef __attribute__((ext_vector_type(8)))  float    v8f;

#define CCH   128      // channels
#define NPTS  10000
#define QPTS  8192
#define KNBR  64
#define QS    66       // padded row stride for query scores in LDS

// ---- WMMA 16-bit fragment address helpers (ISA 7.12.2) ----------------------
// A (16x32 f16): lane = (m&15) + 16*half, half=(kk>>3)&1, elem e packs K:
//   e = (kk&7) + (kk>=16 ? 8 : 0). B (32x16) mirrors with lane = n-column.
__device__ __forceinline__ int frag_half(int kk) { return (kk >> 3) & 1; }
__device__ __forceinline__ int frag_e(int kk)    { return (kk & 7) + ((kk & 16) ? 8 : 0); }

// ---- prep: convert f32 weight [O][Cin] -> f16 B-fragment-major --------------
// dst layout: [kstep][ntile][lane(32)][e(16)] halves; zero-padded K beyond Cin.
__global__ void prep_weights(const float* __restrict__ W, _Float16* __restrict__ dst,
                             int Cin, int NT, int ksteps) {
    int K = ksteps * 32, N = NT * 16;
    int i = blockIdx.x * blockDim.x + threadIdx.x;
    if (i >= K * N) return;
    int k = i % K, n = i / K;
    float v = (k < Cin) ? W[n * Cin + k] : 0.0f;
    int ks = k >> 5, kk = k & 31;
    int lane = (n & 15) + (frag_half(kk) << 4);
    int nt   = n >> 4;
    dst[(((ks * NT) + nt) * 32 + lane) * 16 + frag_e(kk)] = (_Float16)v;
}

// ---- GEMM over fragments: A from LDS, B from global (L2-resident) -----------
template<int NFR>
__device__ __forceinline__ void gemm_frag(const _Float16* __restrict__ aB, int aStride,
                                          const v16h* __restrict__ wf, int NT,
                                          int ksteps, int Mt, int ntb, int lane, v8f* acc) {
    for (int ks = 0; ks < ksteps; ++ks) {
        const v16h a = *(const v16h*)(aB + (((Mt * aStride) + ks) * 32 + lane) * 16);
#pragma unroll
        for (int j = 0; j < NFR; ++j) {
            const v16h b = wf[((ks * NT) + (ntb + j)) * 32 + lane];
            acc[j] = __builtin_amdgcn_wmma_f32_16x16x32_f16(
                false, a, false, b, (short)0, acc[j], false, false);
        }
    }
}

// ---- C-fragment -> next layer's A-fragment (bias + optional ReLU, f16) ------
__device__ __forceinline__ void store_act(_Float16* __restrict__ outB, int outStride,
                                          int Mt, int nt, int lane, const v8f acc,
                                          float bias, bool relu) {
#pragma unroll
    for (int r = 0; r < 8; ++r) {
        float v = acc[r] + bias;
        if (relu) v = fmaxf(v, 0.0f);
        int m  = r + ((lane >> 4) << 3);          // row inside M-tile (C-frag layout)
        int k  = nt * 16 + (lane & 15);           // output channel = next K index
        int ks = k >> 5, kk = k & 31;
        int laneA = m + (frag_half(kk) << 4);
        outB[(((Mt * outStride) + ks) * 32 + laneA) * 16 + frag_e(kk)] = (_Float16)v;
    }
}

// ---- main kernel: one block per query point q -------------------------------
__global__ __launch_bounds__(256)
void interp_attn_main(const float* __restrict__ latents, const float* __restrict__ pos,
                      const float* __restrict__ posnm,   const int* __restrict__ proj,
                      const _Float16* __restrict__ wsf,
                      const float* __restrict__ b1, const float* __restrict__ b2,
                      const float* __restrict__ b3, const float* __restrict__ bq,
                      const float* __restrict__ bv, const float* __restrict__ w8,
                      const float* __restrict__ b8, float* __restrict__ out) {
    const int q    = blockIdx.x;
    const int tid  = threadIdx.x;
    const int lane = tid & 31;
    const int w    = tid >> 5;
    const int Mt   = w >> 1;            // M-tile (rows 16*Mt..16*Mt+15)
    const int ntb4 = (w & 1) * 4;       // N-tile base for 8-ntile GEMMs

    __shared__ __align__(32) _Float16 aBuf0[4 * 5 * 32 * 16]; // X0/x2 (kstride 5)
    __shared__ __align__(32) _Float16 aBuf1[4 * 4 * 32 * 16]; // x1/x3 (kstride 4)
    __shared__ float qSc[64 * QS];
    __shared__ float featS[CCH];
    __shared__ float attnS[KNBR];
    __shared__ float mxS[KNBR];
    __shared__ float smS[KNBR];
    __shared__ int   idxS[KNBR];

    const v16h* w1f = (const v16h*)(wsf);            // 5 ks x 8 nt
    const v16h* w2f = (const v16h*)(wsf + 20480);    // 4 ks x 8 nt
    const v16h* w3f = (const v16h*)(wsf + 36864);    // 4 ks x 8 nt
    const v16h* wqf = (const v16h*)(wsf + 53248);    // 4 ks x 4 nt
    const v16h* wvf = (const v16h*)(wsf + 61440);    // 4 ks x 8 nt

    // phase A: indices, zero pad-kstep of X0, zero feat accumulator
    if (tid < KNBR)  idxS[tid]  = proj[q * KNBR + tid];
    if (tid < CCH)   featS[tid] = 0.0f;
    for (int i = tid; i < 4 * 512; i += 256) {            // kstep 4 = cols 128..159
        int mt = i >> 9, off = i & 511;
        aBuf0[((mt * 5 + 4) * 512) + off] = (_Float16)0.0f;
    }
    __syncthreads();

    // phase B: relative positions (cols 128..130) + latent gather (cols 0..127)
    if (tid < 192) {
        int k = tid & 63, d = tid >> 6;
        float r = posnm[d * QPTS + q] - pos[d * NPTS + idxS[k]];
        int mt = k >> 4;
        aBuf0[(((mt * 5 + 4) * 32 + (k & 15)) * 16) + d] = (_Float16)r; // kk=d<8 -> half0,e=d
    }
    for (int i = tid; i < KNBR * CCH; i += 256) {
        int m = i & 63, c = i >> 6;
        float v = latents[c * NPTS + idxS[m]];
        int mt = m >> 4, ks = c >> 5, kk = c & 31;
        int laneA = (m & 15) + (frag_half(kk) << 4);
        aBuf0[(((mt * 5 + ks) * 32 + laneA) * 16) + frag_e(kk)] = (_Float16)v;
    }
    __syncthreads();

    // layer 1: [64x160] x [160x128] -> aBuf1
    {
        v8f acc[4] = {};
        gemm_frag<4>(aBuf0, 5, w1f, 8, 5, Mt, ntb4, lane, acc);
#pragma unroll
        for (int j = 0; j < 4; ++j)
            store_act(aBuf1, 4, Mt, ntb4 + j, lane, acc[j],
                      b1[(ntb4 + j) * 16 + (lane & 15)], true);
        __syncthreads();
    }
    // layer 2: -> aBuf0 (reuse, kstride 5, ks 0..3 used)
    {
        v8f acc[4] = {};
        gemm_frag<4>(aBuf1, 4, w2f, 8, 4, Mt, ntb4, lane, acc);
#pragma unroll
        for (int j = 0; j < 4; ++j)
            store_act(aBuf0, 5, Mt, ntb4 + j, lane, acc[j],
                      b2[(ntb4 + j) * 16 + (lane & 15)], true);
        __syncthreads();
    }
    // layer 3: -> aBuf1 (x3)
    {
        v8f acc[4] = {};
        gemm_frag<4>(aBuf0, 5, w3f, 8, 4, Mt, ntb4, lane, acc);
#pragma unroll
        for (int j = 0; j < 4; ++j)
            store_act(aBuf1, 4, Mt, ntb4 + j, lane, acc[j],
                      b3[(ntb4 + j) * 16 + (lane & 15)], true);
        __syncthreads();
    }
    // query: [64x128] x [128x64] -> qSc (2 N-tiles per wave)
    {
        v8f acc[2] = {};
        int ntb2 = (w & 1) * 2;
        gemm_frag<2>(aBuf1, 4, wqf, 4, 4, Mt, ntb2, lane, acc);
#pragma unroll
        for (int j = 0; j < 2; ++j) {
            int nt = ntb2 + j;
            float bb = bq[nt * 16 + (lane & 15)];
#pragma unroll
            for (int r = 0; r < 8; ++r) {
                int row = Mt * 16 + r + ((lane >> 4) << 3);
                qSc[row * QS + nt * 16 + (lane & 15)] = acc[j][r] + bb;
            }
        }
        __syncthreads();
    }
    // softmax over neighbors per head, then mean over heads -> attnS[row]
    if (tid < 64) {                        // one head per thread
        float mx = -1e30f;
        for (int r = 0; r < 64; ++r) mx = fmaxf(mx, qSc[r * QS + tid]);
        float s = 0.0f;
        for (int r = 0; r < 64; ++r) s += __expf(qSc[r * QS + tid] - mx);
        mxS[tid] = mx; smS[tid] = s;
    }
    __syncthreads();
    if (tid < 64) {                        // one row per thread
        float a = 0.0f;
        for (int h = 0; h < 64; ++h) a += __expf(qSc[tid * QS + h] - mxS[h]) / smS[h];
        attnS[tid] = a * (1.0f / 64.0f);
    }
    __syncthreads();
    // value GEMM + attention-weighted reduce over neighbors
    {
        v8f acc[4] = {};
        gemm_frag<4>(aBuf1, 4, wvf, 8, 4, Mt, ntb4, lane, acc);
#pragma unroll
        for (int j = 0; j < 4; ++j) {
            int nt = ntb4 + j;
            float bb = bv[nt * 16 + (lane & 15)];
            float partial = 0.0f;
#pragma unroll
            for (int r = 0; r < 8; ++r) {
                int row = Mt * 16 + r + ((lane >> 4) << 3);
                partial += attnS[row] * (acc[j][r] + bb);
            }
            atomicAdd(&featS[nt * 16 + (lane & 15)], partial);
        }
        __syncthreads();
    }
    // final 2x128 projection
    if (tid < 2) {
        float s = b8[tid];
        for (int c = 0; c < CCH; ++c) s += w8[tid * CCH + c] * featS[c];
        out[tid * QPTS + q] = s;
    }
}

extern "C" void kernel_launch(void* const* d_in, const int* in_sizes, int n_in,
                              void* d_out, int out_size, void* d_ws, size_t ws_size,
                              hipStream_t stream) {
    (void)in_sizes; (void)n_in; (void)out_size; (void)ws_size;
    const float* latents = (const float*)d_in[0];
    const float* pos     = (const float*)d_in[1];
    const float* posnm   = (const float*)d_in[2];
    const int*   proj    = (const int*)d_in[3];
    const float* w1 = (const float*)d_in[4];  const float* b1 = (const float*)d_in[5];
    const float* w2 = (const float*)d_in[6];  const float* b2 = (const float*)d_in[7];
    const float* w3 = (const float*)d_in[8];  const float* b3 = (const float*)d_in[9];
    const float* wq = (const float*)d_in[10]; const float* bq = (const float*)d_in[11];
    const float* wv = (const float*)d_in[12]; const float* bv = (const float*)d_in[13];
    const float* w8 = (const float*)d_in[14]; const float* b8 = (const float*)d_in[15];
    _Float16* wsf = (_Float16*)d_ws;   // 155,648 bytes of fragment-major f16 weights

    auto prep = [&](const float* W, _Float16* dst, int Cin, int NT, int ksteps) {
        int total = ksteps * 32 * NT * 16;
        prep_weights<<<(total + 255) / 256, 256, 0, stream>>>(W, dst, Cin, NT, ksteps);
    };
    prep(w1, wsf + 0,     131, 8, 5);   // 20480 halves
    prep(w2, wsf + 20480, 128, 8, 4);   // 16384
    prep(w3, wsf + 36864, 128, 8, 4);   // 16384
    prep(wq, wsf + 53248, 128, 4, 4);   //  8192
    prep(wv, wsf + 61440, 128, 8, 4);   // 16384

    interp_attn_main<<<QPTS, 256, 0, stream>>>(latents, pos, posnm, proj, wsf,
                                               b1, b2, b3, bq, bv, w8, b8,
                                               (float*)d_out);
}
